// WindowMSA_7404523618522
// MI455X (gfx1250) — matrix operand order repaired and checked by reading, not compiled, not run on hardware
//
#include <hip/hip_runtime.h>
#include <hip/hip_bf16.h>

typedef __attribute__((ext_vector_type(16))) _Float16 v16h;
typedef __attribute__((ext_vector_type(8)))  _Float16 v8h;
typedef __attribute__((ext_vector_type(4)))  _Float16 v4h;
typedef __attribute__((ext_vector_type(8)))  float    v8f;

#define NTOK  49
#define NP    64
#define CDIM  128
#define HEADS 4
#define DHEAD 32
#define ATT_SCALE 0.17677669529663687f

// LDS row strides (in elements), 16B-aligned + bank-conflict friendly
#define XB_S 136   // x / y buffer (halves): 272B rows
#define QK_S 40    // q,k rows (halves): 80B
#define VT_S 72    // vT rows (halves): 144B
#define S_S  65    // S rows (floats)
#define P_S  72    // P rows (halves): 144B

__device__ __forceinline__ v16h ldfrag(const _Float16* p0, const _Float16* p1) {
  v8h a = *(const v8h*)p0;
  v8h b = *(const v8h*)p1;
  return __builtin_shufflevector(a, b, 0,1,2,3,4,5,6,7,8,9,10,11,12,13,14,15);
}

__device__ __forceinline__ v8f wmma_f16(v16h a, v16h b, v8f c) {
  return __builtin_amdgcn_wmma_f32_16x16x32_f16(false, a, false, b, (short)0, c,
                                                false, false);
}

// ---------------- prepass: f32 -> f16 weight conversion ----------------
__global__ void convert_weights(const float* __restrict__ qkv_w,
                                const float* __restrict__ proj_w,
                                _Float16* __restrict__ wq,
                                _Float16* __restrict__ wp) {
  int i = blockIdx.x * blockDim.x + threadIdx.x;
  if (i < 3 * CDIM * CDIM) {
    wq[i] = (_Float16)qkv_w[i];
  } else {
    int j = i - 3 * CDIM * CDIM;
    if (j < CDIM * CDIM) wp[j] = (_Float16)proj_w[j];
  }
}

// ---------------- main kernel: one block = one window ----------------
__global__ __launch_bounds__(256)
void win_msa(const float* __restrict__ x, const float* __restrict__ mask,
             const float* __restrict__ qkv_b, const float* __restrict__ proj_b,
             const float* __restrict__ bias_table,
             const _Float16* __restrict__ wq, const _Float16* __restrict__ wp,
             float* __restrict__ out) {
  __shared__ __align__(16) _Float16 sm_xby[NP * XB_S];   // x (f16), later y
  __shared__ __align__(16) _Float16 sm_q[NP * QK_S];     // per-head q (m,d)
  __shared__ __align__(16) _Float16 sm_k[NP * QK_S];     // per-head k (m,d)
  __shared__ __align__(16) _Float16 sm_vT[DHEAD * VT_S]; // per-head v^T (d,m)
  __shared__ __align__(16) float    sm_S[NP * S_S];      // per-head scores f32
  __shared__ __align__(16) _Float16 sm_P[NP * P_S];      // per-head probs f16
  __shared__ int sm_coords[NTOK];

  const int tid  = threadIdx.x;
  const int wave = __builtin_amdgcn_readfirstlane(tid >> 5);  // uniform wave id
  const int lane = tid & 31;
  const int hlf  = lane >> 4;   // lane half (0/1)
  const int l16  = lane & 15;
  const int b    = blockIdx.x;
  const int wi   = b & 63;      // mask window index (nW = 64)

  // ---- phase 1: stage x into LDS as f16, zero-pad rows 49..63 ----
  if (tid < NTOK) sm_coords[tid] = 13 * (tid / 7) + (tid % 7);
  const float* xg = x + (size_t)b * NTOK * CDIM;
  for (int it = tid; it < NP * (CDIM / 4); it += 256) {
    int row = it >> 5;
    int c4  = (it & 31) * 4;
    float4 v = make_float4(0.f, 0.f, 0.f, 0.f);
    if (row < NTOK) v = *(const float4*)(xg + row * CDIM + c4);
    v4h h4;
    h4[0] = (_Float16)v.x; h4[1] = (_Float16)v.y;
    h4[2] = (_Float16)v.z; h4[3] = (_Float16)v.w;
    *(v4h*)(sm_xby + row * XB_S + c4) = h4;
  }
  __syncthreads();

  v8f oacc[HEADS];
  const int rt_o = wave >> 1;   // output row tile for P*v
  const int dt_o = wave & 1;    // output d tile for P*v

  // fixed per-wave tile assignments (enable operand reuse)
  const int rt_qkv = wave >> 1;          // QKV: fixed row tile
  const int ctb    = (wave & 1) * 3;     // QKV: 3 column tiles
  const int ct_s   = wave >> 1;          // S: fixed column tile (k tile)
  const int rtb_s  = (wave & 1) * 2;     // S: 2 row tiles

#pragma unroll
  for (int h = 0; h < HEADS; ++h) {
    // ---- phase 2: per-head QKV GEMM; A fragments hoisted per wave ----
    {
      v16h a4[4];
#pragma unroll
      for (int kk = 0; kk < 4; ++kk) {
        const _Float16* ap =
            sm_xby + (rt_qkv * 16 + l16) * XB_S + kk * 32 + hlf * 8;
        a4[kk] = ldfrag(ap, ap + 16);
      }
      const int m0 = rt_qkv * 16 + 8 * hlf;
#pragma unroll
      for (int i = 0; i < 3; ++i) {
        int ct  = ctb + i;                 // 0..5 (uniform)
        int sec = ct >> 1;                 // 0=q 1=k 2=v (uniform)
        int d0  = (ct & 1) * 16;
        int j   = sec * CDIM + h * DHEAD + d0 + l16;  // weight row = out col
        v8f acc = {};
#pragma unroll
        for (int kk = 0; kk < 4; ++kk) {
          const _Float16* bp = wq + j * CDIM + kk * 32 + hlf * 8;
          acc = wmma_f16(a4[kk], ldfrag(bp, bp + 16), acc);
        }
        float bj = qkv_b[j];
        _Float16* p0;
        int step;
        if (sec == 0)      { p0 = sm_q  + m0 * QK_S + d0 + l16; step = QK_S; }
        else if (sec == 1) { p0 = sm_k  + m0 * QK_S + d0 + l16; step = QK_S; }
        else               { p0 = sm_vT + (d0 + l16) * VT_S + m0; step = 1;  }
#pragma unroll
        for (int r = 0; r < 8; ++r) p0[r * step] = (_Float16)(acc[r] + bj);
      }
    }
    __syncthreads();

    // ---- phase 3: S = scale * q k^T (shared k fragment per wave) ----
    {
      const _Float16* bp = sm_k + (ct_s * 16 + l16) * QK_S + hlf * 8;
      v16h kf = ldfrag(bp, bp + 16);
#pragma unroll
      for (int i = 0; i < 2; ++i) {
        int rt = rtb_s + i;
        const _Float16* ap = sm_q + (rt * 16 + l16) * QK_S + hlf * 8;
        v8f acc = {};
        acc = wmma_f16(ldfrag(ap, ap + 16), kf, acc);
#pragma unroll
        for (int r = 0; r < 8; ++r)
          sm_S[(rt * 16 + r + 8 * hlf) * S_S + ct_s * 16 + l16] =
              acc[r] * ATT_SCALE;
      }
    }
    __syncthreads();

    // ---- phase 4: softmax rows with rel-bias + shift mask -> P (f16) ----
    if (tid < NTOK) {
      int n  = tid;
      int cn = sm_coords[n];
      const float* mrow = mask + (size_t)wi * NTOK * NTOK + n * NTOK;
      float* srow = sm_S + n * S_S;
      float mx = -1e30f;
      for (int m = 0; m < NTOK; ++m) {
        int idx = cn + sm_coords[48 - m];                 // REL_IDX[n][m]
        float v = srow[m] + bias_table[idx * HEADS + h] + mrow[m];
        srow[m] = v;
        mx = fmaxf(mx, v);
      }
      float sum = 0.f;
      for (int m = 0; m < NTOK; ++m) {
        float e = __expf(srow[m] - mx);
        srow[m] = e;
        sum += e;
      }
      float inv = 1.f / sum;
      _Float16* prow = sm_P + n * P_S;
      for (int m = 0; m < NTOK; ++m) prow[m] = (_Float16)(srow[m] * inv);
      for (int m = NTOK; m < NP; ++m) prow[m] = (_Float16)0.f;
    } else if (tid < NP) {          // zero padded rows 49..63
      _Float16* prow = sm_P + tid * P_S;
      for (int m = 0; m < NP; ++m) prow[m] = (_Float16)0.f;
    }
    __syncthreads();

    // ---- phase 5: O_h = P * v  (per wave: one 16x16 tile, K=64) ----
    {
      v8f acc = {};
#pragma unroll
      for (int kk = 0; kk < 2; ++kk) {
        const _Float16* ap = sm_P + (rt_o * 16 + l16) * P_S + kk * 32 + hlf * 8;
        const _Float16* bp =
            sm_vT + (dt_o * 16 + l16) * VT_S + kk * 32 + hlf * 8;
        acc = wmma_f16(ldfrag(ap, ap + 16), ldfrag(bp, bp + 16), acc);
      }
      oacc[h] = acc;
    }
    __syncthreads();  // next head reuses q/k/vT/S/P
  }

  // ---- spill O (all heads) into y, overlaying dead x buffer ----
#pragma unroll
  for (int h = 0; h < HEADS; ++h) {
#pragma unroll
    for (int r = 0; r < 8; ++r)
      sm_xby[(rt_o * 16 + r + 8 * hlf) * XB_S + h * DHEAD + dt_o * 16 + l16] =
          (_Float16)oacc[h][r];
  }
  __syncthreads();

  // ---- phase 6: out = y proj_w^T + proj_b (B fragments hoisted) ----
  {
    const int ct = wave;             // 8 col tiles, one per wave
    v16h b4[4];
#pragma unroll
    for (int kk = 0; kk < 4; ++kk) {
      const _Float16* bp = wp + (ct * 16 + l16) * CDIM + kk * 32 + hlf * 8;
      b4[kk] = ldfrag(bp, bp + 16);
    }
    const int   col = ct * 16 + l16;
    const float bj  = proj_b[col];
    float* ob = out + (size_t)b * NTOK * CDIM;
#pragma unroll
    for (int rt = 0; rt < 4; ++rt) {
      v8f acc = {};
#pragma unroll
      for (int kk = 0; kk < 4; ++kk) {
        const _Float16* ap =
            sm_xby + (rt * 16 + l16) * XB_S + kk * 32 + hlf * 8;
        acc = wmma_f16(ldfrag(ap, ap + 16), b4[kk], acc);
      }
      if (rt < 3) {                  // all 16 rows valid: branch-free stores
#pragma unroll
        for (int r = 0; r < 8; ++r)
          ob[(rt * 16 + r + 8 * hlf) * CDIM + col] = acc[r] + bj;
      } else if (hlf == 0) {         // rt==3: only row 48 is in range
        ob[48 * CDIM + col] = acc[0] + bj;
      }
    }
  }
}

extern "C" void kernel_launch(void* const* d_in, const int* in_sizes, int n_in,
                              void* d_out, int out_size, void* d_ws, size_t ws_size,
                              hipStream_t stream) {
  const float* x      = (const float*)d_in[0];
  const float* mask   = (const float*)d_in[1];
  const float* qkv_w  = (const float*)d_in[2];
  const float* qkv_b  = (const float*)d_in[3];
  const float* proj_w = (const float*)d_in[4];
  const float* proj_b = (const float*)d_in[5];
  const float* bias_t = (const float*)d_in[6];
  float* out = (float*)d_out;

  _Float16* wq = (_Float16*)d_ws;            // 3*128*128 halves
  _Float16* wp = wq + 3 * CDIM * CDIM;       // 128*128 halves

  int total = 3 * CDIM * CDIM + CDIM * CDIM; // 65536
  convert_weights<<<(total + 255) / 256, 256, 0, stream>>>(qkv_w, proj_w, wq, wp);

  int B = in_sizes[0] / (NTOK * CDIM);       // 4096
  win_msa<<<B, 256, 0, stream>>>(x, mask, qkv_b, proj_b, bias_t, wq, wp, out);
}
